// LSTM_10488310137258
// MI455X (gfx1250) — compile-verified, matrix operand order checked by reading, not run on hardware
//
#include <hip/hip_runtime.h>
#include <cstdint>
#include <cstddef>

// Problem constants (match reference)
#define BATCH 1024
#define TSTEPS 256
#define FDIM 7
#define HDIM 100
#define GDIM 400      // 4*H gates (i,f,g,o)
#define NLAYERS 10

#define NTHREADS 256   // 8 wave32 waves
#define NWAVES 8

typedef _Float16 half_t;
typedef __attribute__((ext_vector_type(16))) _Float16 v16h;
typedef __attribute__((ext_vector_type(8)))  _Float16 v8h;
typedef __attribute__((ext_vector_type(4)))  _Float16 v4h;
typedef __attribute__((ext_vector_type(2)))  _Float16 v2h;
typedef __attribute__((ext_vector_type(8)))  float    v8f;
typedef __attribute__((ext_vector_type(4)))  float    v4f;

// ---- LDS layout (bytes). Padded strides chosen for conflict-free access. ----
constexpr int KS = 136;                       // K stride in f16 (128 padded + 8)
constexpr int ZS = 404;                       // gate stride in f32 (400 + 4)
constexpr int OFF_WHH  = 0;                         // [G][KS] f16   (108800 B)
constexpr int OFF_WIH  = OFF_WHH  + GDIM * KS * 2;  // [G][KS] f16   (108800 B)
constexpr int OFF_XIN  = OFF_WIH  + GDIM * KS * 2;  // [16][KS] f16  (4352 B)
constexpr int OFF_HCUR = OFF_XIN  + 16 * KS * 2;    // [16][KS] f16  (4352 B)
constexpr int OFF_Z    = OFF_HCUR + 16 * KS * 2;    // [16][ZS] f32  (25856 B)
constexpr int OFF_C    = OFF_Z    + 16 * ZS * 4;    // [16][H] f32   (6400 B)
constexpr int OFF_HFIN = OFF_C    + 16 * HDIM * 4;  // [16][H] f32   (6400 B)
constexpr int OFF_BIAS = OFF_HFIN + 16 * HDIM * 4;  // [G] f32       (1600 B)
constexpr int SMEM_BYTES = OFF_BIAS + GDIM * 4;     // ~266 KB < 320 KB/WG

// Build a 16x32 f16 WMMA operand fragment from LDS.
// A-matrix: row = M = lane%16 ; B-matrix: row = N-index = nt*16 + lane%16.
// Lanes 0-15 hold K = kc*32 + {0..7, 16..23}; lanes 16-31 K = kc*32 + {8..15, 24..31}.
__device__ __forceinline__ v16h load_frag(const half_t* base, int row, int kc, int lane) {
    const int khalf = (lane < 16) ? 0 : 8;
    const half_t* p = base + row * KS + kc * 32 + khalf;
    v8h lo = *(const v8h*)(p);
    v8h hi = *(const v8h*)(p + 16);
    return __builtin_shufflevector(lo, hi, 0,1,2,3,4,5,6,7,8,9,10,11,12,13,14,15);
}

__device__ __forceinline__ float sigf(float x) { return 1.0f / (1.0f + __expf(-x)); }

// One full T-step scan of one LSTM layer. NKIN = number of 32-wide K-chunks of
// the input GEMM (1 for layer 0 with F=7, 4 for H=100 layers). Compile-time so
// every fragment array is constant-indexed (no v_movrels).
template<int NKIN>
__device__ __forceinline__ void time_scan(
    const float* __restrict__ x, const half_t* __restrict__ hs_in,
    half_t* __restrict__ hs_out,
    half_t* whh, half_t* wih, half_t* xin, half_t* hcur,
    float* zbuf, float* cst, float* hfin, const float* bias,
    int tid, int lane, int wave, int b0, bool last_layer)
{
    #pragma unroll 1
    for (int t = 0; t < TSTEPS; ++t) {
        // ---- (a) stage x_t tile into LDS ----
        if (NKIN == 1) {
            for (int i = tid; i < 16 * FDIM; i += NTHREADS) {
                int r = i / FDIM, k = i % FDIM;
                xin[r * KS + k] =
                    (half_t)x[((size_t)(b0 + r) * TSTEPS + t) * FDIM + k];
            }
        } else {
            const half_t* src = hs_in + ((size_t)t * BATCH + b0) * HDIM;
            for (int i = tid; i < 16 * 50; i += NTHREADS) {  // 100 f16 = 50 dwords/row
                int r = i / 50, jp = i % 50;
                *(uint32_t*)(xin + r * KS + jp * 2) =
                    *(const uint32_t*)(src + (size_t)r * HDIM + jp * 2);
            }
            if (t + 1 < TSTEPS && tid < 16)
                __builtin_prefetch(hs_in + ((size_t)(t + 1) * BATCH + b0 + tid) * HDIM, 0, 0);
        }
        __syncthreads();

        // ---- (b) fused GEMM: z = xin @ W_ih^T + h @ W_hh^T  (WMMA f16->f32) ----
        v16h ah[4], ax[NKIN];
        #pragma unroll
        for (int kc = 0; kc < 4; ++kc) ah[kc] = load_frag(hcur, lane & 15, kc, lane);
        #pragma unroll
        for (int kc = 0; kc < NKIN; ++kc) ax[kc] = load_frag(xin, lane & 15, kc, lane);

        #pragma unroll 1
        for (int nt = wave; nt < 25; nt += NWAVES) {   // 25 N-tiles of 16 gates
            const int brow = nt * 16 + (lane & 15);
            // Preload all B fragments so ds_loads overlap the wmma chain.
            v16h bh[4], bx[NKIN];
            #pragma unroll
            for (int kc = 0; kc < 4; ++kc) bh[kc] = load_frag(whh, brow, kc, lane);
            #pragma unroll
            for (int kc = 0; kc < NKIN; ++kc) bx[kc] = load_frag(wih, brow, kc, lane);

            v8f acc = {};
            #pragma unroll
            for (int kc = 0; kc < 4; ++kc)
                acc = __builtin_amdgcn_wmma_f32_16x16x32_f16(
                    false, ah[kc], false, bh[kc], (short)0, acc, false, false);
            #pragma unroll
            for (int kc = 0; kc < NKIN; ++kc)
                acc = __builtin_amdgcn_wmma_f32_16x16x32_f16(
                    false, ax[kc], false, bx[kc], (short)0, acc, false, false);

            const int ncol  = nt * 16 + (lane & 15);
            const int mbase = (lane < 16) ? 0 : 8;
            #pragma unroll
            for (int i = 0; i < 8; ++i) zbuf[(mbase + i) * ZS + ncol] = acc[i];
        }
        __syncthreads();

        // ---- (c) gates + state update (PyTorch gate order i,f,g,o), 2/iter ----
        for (int p = tid; p < 16 * 50; p += NTHREADS) {
            int r = p / 50, j = (p % 50) * 2;
            float hv[2];
            #pragma unroll
            for (int u = 0; u < 2; ++u) {
                int jj = j + u;
                float zi = zbuf[r * ZS + jj]            + bias[jj];
                float zf = zbuf[r * ZS + HDIM + jj]     + bias[HDIM + jj];
                float zg = zbuf[r * ZS + 2 * HDIM + jj] + bias[2 * HDIM + jj];
                float zo = zbuf[r * ZS + 3 * HDIM + jj] + bias[3 * HDIM + jj];
                float c  = sigf(zf) * cst[r * HDIM + jj] + sigf(zi) * tanhf(zg);
                float h  = sigf(zo) * tanhf(c);
                cst[r * HDIM + jj] = c;
                if (last_layer && t == TSTEPS - 1) hfin[r * HDIM + jj] = h;
                hv[u] = h;
            }
            v2h pack = { (half_t)hv[0], (half_t)hv[1] };
            *(v2h*)(hcur + r * KS + j) = pack;
            *(v2h*)(hs_out + ((size_t)t * BATCH + b0 + r) * HDIM + j) = pack;
        }
        __syncthreads();
    }
}

__global__ void __launch_bounds__(NTHREADS)
lstm_persistent_kernel(const float* __restrict__ x,          // [B,T,F]
                       const float* __restrict__ W_ih0,      // [G,F]
                       const float* __restrict__ W_ih_rest,  // [L-1,G,H]
                       const float* __restrict__ W_hh,       // [L,G,H]
                       const float* __restrict__ b_ih,       // [L,G]
                       const float* __restrict__ b_hh,       // [L,G]
                       const float* __restrict__ W_lin,      // [1,H]
                       const float* __restrict__ b_lin,      // [1]
                       float* __restrict__ out,              // [B,1]
                       half_t* __restrict__ hs_a,            // [T,B,H] ping
                       half_t* __restrict__ hs_b)            // [T,B,H] pong
{
    extern __shared__ __align__(16) char smem[];
    half_t* whh  = (half_t*)(smem + OFF_WHH);
    half_t* wih  = (half_t*)(smem + OFF_WIH);
    half_t* xin  = (half_t*)(smem + OFF_XIN);
    half_t* hcur = (half_t*)(smem + OFF_HCUR);
    float*  zbuf = (float*) (smem + OFF_Z);
    float*  cst  = (float*) (smem + OFF_C);
    float*  hfin = (float*) (smem + OFF_HFIN);
    float*  bias = (float*) (smem + OFF_BIAS);

    const int tid  = threadIdx.x;
    const int lane = tid & 31;
    // Force wave index into an SGPR: makes the tile loop scalar (SCC branch)
    // instead of an EXEC-masked divergent loop around the WMMA chain.
    const int wave = __builtin_amdgcn_readfirstlane(tid >> 5);
    const int b0   = blockIdx.x * 16;   // 16-row batch tile, fully independent

    // ---- one-time zero: weight pads + operand pads (never dirtied later) ----
    for (int i = tid; i < GDIM * KS / 2; i += NTHREADS) {
        ((uint32_t*)whh)[i] = 0u;
        ((uint32_t*)wih)[i] = 0u;
    }
    for (int i = tid; i < 16 * KS / 2; i += NTHREADS) {
        ((uint32_t*)xin)[i]  = 0u;
        ((uint32_t*)hcur)[i] = 0u;
    }

    #pragma unroll 1
    for (int l = 0; l < NLAYERS; ++l) {
        // per-layer state reset: h0 = c0 = 0 (active region only)
        for (int p = tid; p < 16 * 50; p += NTHREADS) {
            int r = p / 50, j = (p % 50) * 2;
            *(uint32_t*)(hcur + r * KS + j) = 0u;
            cst[r * HDIM + j]     = 0.0f;
            cst[r * HDIM + j + 1] = 0.0f;
        }
        __syncthreads();

        // ---- load layer weights (f32 global -> f16 LDS, vectorized) ----
        for (int i = tid; i < GDIM * 25; i += NTHREADS) {  // 100 = 25 * f32x4
            int n = i / 25, kq = i % 25;
            v4f w = *(const v4f*)(W_hh + ((size_t)l * GDIM + n) * HDIM + kq * 4);
            v4h h4 = { (half_t)w[0], (half_t)w[1], (half_t)w[2], (half_t)w[3] };
            *(v4h*)(whh + n * KS + kq * 4) = h4;
        }
        if (l == 0) {
            for (int i = tid; i < GDIM * FDIM; i += NTHREADS) {
                int n = i / FDIM, k = i % FDIM;
                wih[n * KS + k] = (half_t)W_ih0[(size_t)n * FDIM + k];
            }
        } else {
            for (int i = tid; i < GDIM * 25; i += NTHREADS) {
                int n = i / 25, kq = i % 25;
                v4f w = *(const v4f*)(W_ih_rest + ((size_t)(l - 1) * GDIM + n) * HDIM + kq * 4);
                v4h h4 = { (half_t)w[0], (half_t)w[1], (half_t)w[2], (half_t)w[3] };
                *(v4h*)(wih + n * KS + kq * 4) = h4;
            }
        }
        for (int i = tid; i < GDIM; i += NTHREADS)
            bias[i] = b_ih[(size_t)l * GDIM + i] + b_hh[(size_t)l * GDIM + i];
        __syncthreads();

        const half_t* hs_in  = (l == 0) ? nullptr : (((l & 1) == 1) ? hs_a : hs_b);
        half_t*       hs_out = ((l & 1) == 0) ? hs_a : hs_b;
        const bool    last   = (l == NLAYERS - 1);

        if (l == 0)
            time_scan<1>(x, hs_in, hs_out, whh, wih, xin, hcur,
                         zbuf, cst, hfin, bias, tid, lane, wave, b0, last);
        else
            time_scan<4>(x, hs_in, hs_out, whh, wih, xin, hcur,
                         zbuf, cst, hfin, bias, tid, lane, wave, b0, last);
    }

    // ---- final linear: out[b] = h_last . W_lin + b_lin ----
    if (tid < 16) {
        float acc = b_lin[0];
        #pragma unroll 4
        for (int j = 0; j < HDIM; ++j) acc += hfin[tid * HDIM + j] * W_lin[j];
        out[b0 + tid] = acc;
    }
}

extern "C" void kernel_launch(void* const* d_in, const int* in_sizes, int n_in,
                              void* d_out, int out_size, void* d_ws, size_t ws_size,
                              hipStream_t stream) {
    (void)in_sizes; (void)n_in; (void)out_size; (void)ws_size;
    const float* x         = (const float*)d_in[0];
    const float* W_ih0     = (const float*)d_in[1];
    const float* W_ih_rest = (const float*)d_in[2];
    const float* W_hh      = (const float*)d_in[3];
    const float* b_ih      = (const float*)d_in[4];
    const float* b_hh      = (const float*)d_in[5];
    const float* W_lin     = (const float*)d_in[6];
    const float* b_lin     = (const float*)d_in[7];

    // Workspace: two f16 hidden-sequence buffers [T,B,H] (~52 MB each, L2-resident)
    half_t* hs_a = (half_t*)d_ws;
    half_t* hs_b = hs_a + (size_t)TSTEPS * BATCH * HDIM;

    dim3 grid(BATCH / 16);    // 64 independent workgroups (batch tiles)
    dim3 block(NTHREADS);     // 8 wave32 waves -> critical path 4 tiles/wave
    hipLaunchKernelGGL(lstm_persistent_kernel, grid, block, SMEM_BYTES, stream,
                       x, W_ih0, W_ih_rest, W_hh, b_ih, b_hh, W_lin, b_lin,
                       (float*)d_out, hs_a, hs_b);
}
